// AntecedentGenerator_16595753631977
// MI455X (gfx1250) — matrix operand order repaired and checked by reading, not compiled, not run on hardware
//
#include <hip/hip_runtime.h>
#include <hip/hip_bf16.h>
#include <math.h>

// ---------------------------------------------------------------------------
// Problem constants (match reference)
// ---------------------------------------------------------------------------
namespace {
constexpr int B_ = 1024;
constexpr int DIN_ = 512;
constexpr int H_ = 768;
constexpr int A_ = 5001;
constexpr int AP_ = 5056;  // A_ rounded up to 64 (padded N for head_W)
constexpr int L_ = 4;
constexpr int HEADS_ = 2;
constexpr int C_ = 2;
constexpr float NDATA_ = 56000.0f;
constexpr float NEG_ = -1e30f;
}

typedef __attribute__((ext_vector_type(16))) __bf16 v16bf;
typedef __attribute__((ext_vector_type(8))) float v8f;
typedef __attribute__((ext_vector_type(4))) unsigned int tdm_v4u;
typedef __attribute__((ext_vector_type(8))) int tdm_v8i;
typedef __attribute__((ext_vector_type(4))) int tdm_v4i;

struct Q2 { uint4 a, b; };  // 32 bytes == one v16bf fragment

__device__ __forceinline__ unsigned short f2bf(float f) {
  // round-to-nearest-even fp32 -> bf16 (bit-level)
  unsigned int u = __float_as_uint(f);
  unsigned int r = u + 0x7FFFu + ((u >> 16) & 1u);
  return (unsigned short)(r >> 16);
}

// ---------------------------------------------------------------------------
// TDM 2D tile load: rows x 32 bf16 elements, contiguous k per row, into LDS
// with pad_interval=16 DWORDs + pad_amount=4 DWORDs (row stride 40 elements).
// D# layout per CDNA5 ISA 8.3/8.4.
// ---------------------------------------------------------------------------
__device__ __forceinline__ void tdm_load_tile(const unsigned short* gaddr,
                                              unsigned lds_off,
                                              int dim0 /*tensor k extent*/,
                                              int dim1 /*tensor row extent*/,
                                              int tileRows) {
  unsigned long long ga = (unsigned long long)(uintptr_t)gaddr;
  tdm_v4u g0;
  g0.x = 1u;                                              // count=1
  g0.y = lds_off;                                         // LDS byte address
  g0.z = (unsigned)(ga & 0xFFFFFFFFu);                    // global addr lo
  g0.w = (unsigned)((ga >> 32) & 0x01FFFFFFu) | (2u << 30);  // addr hi | type=2
  // data_size=2B | pad_enable | pad_interval=16DW (code 3) | pad_amount=4DW (3)
  unsigned w0 = (1u << 16) | (1u << 20) | (3u << 22) | (3u << 25);
  unsigned td0 = (unsigned)dim0, td1 = (unsigned)dim1;
  tdm_v8i g1;
  g1[0] = (int)w0;
  g1[1] = (int)((td0 & 0xFFFFu) << 16);                            // dim0 lo16
  g1[2] = (int)(((td0 >> 16) & 0xFFFFu) | ((td1 & 0xFFFFu) << 16));
  g1[3] = (int)(((td1 >> 16) & 0xFFFFu) | (32u << 16));            // tile_dim0=32
  g1[4] = (int)((unsigned)tileRows);                               // tile_dim1
  g1[5] = (int)td0;                                                // stride0=dim0
  g1[6] = 0;
  g1[7] = 0;
  tdm_v4i gz = {0, 0, 0, 0};
#if defined(__clang_major__) && __clang_major__ >= 23
  tdm_v8i gz8 = {0, 0, 0, 0, 0, 0, 0, 0};
  __builtin_amdgcn_tensor_load_to_lds(g0, g1, gz, gz, gz8, 0);
#else
  __builtin_amdgcn_tensor_load_to_lds(g0, g1, gz, gz, 0);
#endif
}

// ---------------------------------------------------------------------------
// Tiled bf16 WMMA GEMM: C[M,N] = act(A[M,K] @ W^T + bias)
//   A  : bf16 [M,K] activations (row-major)
//   Bw : bf16 [NP,K] pre-transposed weights (k contiguous, NP >= N zero rows)
//   Both tiles staged by the Tensor Data Mover, double-buffered in LDS.
//   Block = 256 threads = 8 wave32; tile = 256(M) x 64(N); K step = 32.
//   Each wave owns a 32-row slab: 2 A fragments feed 8 WMMAs per K step.
//   Assumes: M % 256 == 0, K % 32 == 0, NP % 64 == 0.
// ---------------------------------------------------------------------------
__global__ __launch_bounds__(256) void gemm_wmma_kernel(
    int M, int N, int K,
    const unsigned short* __restrict__ A,
    const unsigned short* __restrict__ Bw, int NP,
    const float* __restrict__ bias,
    float* __restrict__ Cf, unsigned short* __restrict__ Cb,
    int act) {
  __shared__ __align__(16) unsigned short As[2][256][40];  // [buf][m][k]
  __shared__ __align__(16) unsigned short Bt[2][64][40];   // [buf][n][k]

  const int nBase = blockIdx.x * 64;
  const int mBase = blockIdx.y * 256;
  const int tid = threadIdx.x;
  const int wave = tid >> 5;
  const int lane = tid & 31;
  const int lr = lane & 15;   // lane % 16
  const int hi = lane >> 4;   // lane half
  const int m0 = wave * 32;   // wave's 32-row slab within tile

  const unsigned short* Arow = A + (size_t)mBase * K;
  const unsigned short* Brow = Bw + (size_t)nBase * K;

  v8f acc0[4], acc1[4];
  #pragma unroll
  for (int i = 0; i < 4; ++i) {
    acc0[i] = (v8f){0.f, 0.f, 0.f, 0.f, 0.f, 0.f, 0.f, 0.f};
    acc1[i] = (v8f){0.f, 0.f, 0.f, 0.f, 0.f, 0.f, 0.f, 0.f};
  }

  const int nTiles = K >> 5;

  // prologue: DMA tile 0 into buffer 0
  if (wave == 0)
    tdm_load_tile(Arow, (unsigned)(uintptr_t)&As[0][0][0], K, M, 256);
  else if (wave == 1)
    tdm_load_tile(Brow, (unsigned)(uintptr_t)&Bt[0][0][0], K, NP, 64);

  for (int t = 0; t < nTiles; ++t) {
    const int cur = t & 1, nxt = cur ^ 1;
    // issue next tile into the other buffer (its readers finished at the
    // barrier that ended iteration t-1), then wait for tile t.
    if (t + 1 < nTiles) {
      int koff = (t + 1) * 32;
      if (wave == 0)
        tdm_load_tile(Arow + koff, (unsigned)(uintptr_t)&As[nxt][0][0], K, M, 256);
      else if (wave == 1)
        tdm_load_tile(Brow + koff, (unsigned)(uintptr_t)&Bt[nxt][0][0], K, NP, 64);
      if (wave < 2) __builtin_amdgcn_s_wait_tensorcnt(1);  // in-order => t done
    } else {
      if (wave < 2) __builtin_amdgcn_s_wait_tensorcnt(0);
    }
    // warm L2 two tiles ahead (global_prefetch on gfx1250)
    if (t + 2 < nTiles) {
      size_t koff2 = (size_t)(t + 2) * 32;
      if (wave == 2)
        __builtin_prefetch(Arow + (size_t)(lane * 8) * K + koff2, 0, 1);
      else if (wave == 3)
        __builtin_prefetch(Brow + (size_t)(lane * 2) * K + koff2, 0, 1);
    }
    __syncthreads();  // publish tile t (TDM writes) to all waves

    // ---- load ALL fragments first (lets the scheduler pipeline DS waits) ----
    // A fragment (ISA 7.12.2 16x32 bf16): per-lane contiguous runs:
    //   elems 0..7  -> k = 8*hi .. 8*hi+7 ; elems 8..15 -> k = 16+8*hi ..
    Q2 qa0, qa1;
    {
      const uint4* ap0 = (const uint4*)(&As[cur][m0 + lr][0]);
      const uint4* ap1 = (const uint4*)(&As[cur][m0 + 16 + lr][0]);
      qa0.a = ap0[hi];     qa0.b = ap0[2 + hi];
      qa1.a = ap1[hi];     qa1.b = ap1[2 + hi];
    }
    Q2 qb[4];
    #pragma unroll
    for (int ct = 0; ct < 4; ++ct) {
      // B fragment: n = ct*16 + lane%16, k = 8*hi..8*hi+7 and 16+8*hi..16+8*hi+7
      const uint4* bp = (const uint4*)(&Bt[cur][ct * 16 + lr][0]);
      qb[ct].a = bp[hi];
      qb[ct].b = bp[2 + hi];
    }
    v16bf a0 = __builtin_bit_cast(v16bf, qa0);
    v16bf a1 = __builtin_bit_cast(v16bf, qa1);
    #pragma unroll
    for (int ct = 0; ct < 4; ++ct) {
      v16bf bfrag = __builtin_bit_cast(v16bf, qb[ct]);
      acc0[ct] = __builtin_amdgcn_wmma_f32_16x16x32_bf16(
          false, a0, false, bfrag, (short)0, acc0[ct], false, false);
      acc1[ct] = __builtin_amdgcn_wmma_f32_16x16x32_bf16(
          false, a1, false, bfrag, (short)0, acc1[ct], false, false);
    }
    __syncthreads();  // compute done before buffer is re-filled
  }

  // Epilogue: C/D layout — VGPR e holds row e (lanes 0-15) / e+8 (lanes 16-31),
  // N = lane%16. Two 16-row slabs per wave.
  #pragma unroll
  for (int s = 0; s < 2; ++s) {
    #pragma unroll
    for (int ct = 0; ct < 4; ++ct) {
      #pragma unroll
      for (int e = 0; e < 8; ++e) {
        int row = mBase + m0 + s * 16 + e + 8 * hi;
        int col = nBase + ct * 16 + lr;
        if (row < M && col < N) {
          float v = (s == 0) ? acc0[ct][e] : acc1[ct][e];
          if (bias) v += bias[col];
          if (act == 1) v = fmaxf(v, 0.f);
          if (Cf) Cf[(size_t)row * N + col] = v;
          if (Cb) Cb[(size_t)row * N + col] = f2bf(v);
        }
      }
    }
  }
}

// fp32 -> bf16 plain cast (for activations)
__global__ void cvt_bf16_kernel(const float* __restrict__ src,
                                unsigned short* __restrict__ dst, int n) {
  int t = blockIdx.x * blockDim.x + threadIdx.x;
  if (t < n) dst[t] = f2bf(src[t]);
}

// fp32 W[K][N] -> bf16 Wt[NP][K] (transpose + pad rows [N,NP) with zeros).
// LDS-tiled 32x32 so both global sides are coalesced.
__global__ __launch_bounds__(256) void cvt_transpose_kernel(
    const float* __restrict__ src, unsigned short* __restrict__ dst,
    int K, int N, int NP) {
  __shared__ float tile[32][33];
  const int kb = blockIdx.x * 32;
  const int nb = blockIdx.y * 32;
  const int tx = threadIdx.x & 31;
  const int ty = threadIdx.x >> 5;  // 0..7
  #pragma unroll
  for (int i = 0; i < 4; ++i) {
    int k = kb + ty + 8 * i, n = nb + tx;
    tile[ty + 8 * i][tx] = (k < K && n < N) ? src[(size_t)k * N + n] : 0.f;
  }
  __syncthreads();
  #pragma unroll
  for (int i = 0; i < 4; ++i) {
    int n = nb + ty + 8 * i, k = kb + tx;
    if (n < NP && k < K)
      dst[(size_t)n * K + k] = f2bf(tile[tx][ty + 8 * i]);
  }
}

// ---------------------------------------------------------------------------
// GRU gate fusion: h = (1-z)*n + z*h  (torch gate order r,z,n); emits bf16 h too
// ---------------------------------------------------------------------------
__global__ void gru_eltwise_kernel(const float* __restrict__ gi,
                                   const float* __restrict__ gh,
                                   float* __restrict__ h,
                                   unsigned short* __restrict__ hb, int n) {
  int t = blockIdx.x * blockDim.x + threadIdx.x;
  if (t >= n) return;
  int b = t / H_, k = t - b * H_;
  const float* gib = gi + (size_t)b * 3 * H_;
  const float* ghb = gh + (size_t)b * 3 * H_;
  float ir = gib[k],           hr = ghb[k];
  float iz = gib[H_ + k],      hz = ghb[H_ + k];
  float in_ = gib[2 * H_ + k], hn = ghb[2 * H_ + k];
  float r = 1.f / (1.f + expf(-(ir + hr)));
  float z = 1.f / (1.f + expf(-(iz + hz)));
  float nn = tanhf(in_ + r * hn);
  float hv = (1.f - z) * nn + z * h[t];
  h[t] = hv;
  hb[t] = f2bf(hv);
}

// ---------------------------------------------------------------------------
// Deterministic Gumbel noise stand-in (splitmix64 hash -> uniform -> gumbel).
// ---------------------------------------------------------------------------
__device__ __forceinline__ float gumbel_noise(unsigned long long x) {
  x += 0x9E3779B97F4A7C15ull;
  x = (x ^ (x >> 30)) * 0xBF58476D1CE4E5B9ull;
  x = (x ^ (x >> 27)) * 0x94D049BB133111EBull;
  x ^= (x >> 31);
  float u = (float)(x >> 40) * (1.0f / 16777216.0f);
  u = fminf(fmaxf(u, 1e-9f), 0.999999f);
  return -logf(-logf(u));
}

// ---------------------------------------------------------------------------
// Per-row masked Gumbel argmax + one-hot write + mask update (block per row).
// ---------------------------------------------------------------------------
__global__ __launch_bounds__(256) void select_kernel(
    const float* __restrict__ logits, int* __restrict__ mask,
    int* __restrict__ ind, float* __restrict__ atom_probs, int head, int j) {
  const int b = blockIdx.x;
  const int tid = threadIdx.x;
  __shared__ float sv[256];
  __shared__ int si[256];
  int* mrow = mask + (size_t)b * A_;

  if (j == 0) {
    // col0 = (sum(mask)==0) ? 1 : mask[:,0]
    int lsum = 0;
    for (int a = tid; a < A_; a += 256) lsum += mrow[a];
    si[tid] = lsum;
    __syncthreads();
    for (int s = 128; s > 0; s >>= 1) {
      if (tid < s) si[tid] += si[tid + s];
      __syncthreads();
    }
    if (tid == 0 && si[0] == 0) mrow[0] = 1;
    __syncthreads();
  } else {
    int prev = ind[((size_t)head * L_ + (j - 1)) * B_ + b];
    if (prev == 0) {
      for (int a = tid; a < A_; a += 256) mrow[a] = 0;
    }
    __syncthreads();
    if (tid == 0) mrow[0] = 1;
    __syncthreads();
  }

  // argmax over masked logits + gumbel (first-index tie break)
  const float* lrow = logits + (size_t)b * A_;
  float best = -INFINITY;
  int bidx = A_;
  for (int a = tid; a < A_; a += 256) {
    float v;
    if (mrow[a] > 0) {
      unsigned long long gid =
          (((unsigned long long)(head * L_ + j) * B_ + b) * (unsigned long long)A_) + a;
      v = lrow[a] + gumbel_noise(gid);
    } else {
      v = NEG_;
    }
    if (v > best || (v == best && a < bidx)) { best = v; bidx = a; }
  }
  sv[tid] = best; si[tid] = bidx;
  __syncthreads();
  for (int s = 128; s > 0; s >>= 1) {
    if (tid < s) {
      if (sv[tid + s] > sv[tid] ||
          (sv[tid + s] == sv[tid] && si[tid + s] < si[tid])) {
        sv[tid] = sv[tid + s]; si[tid] = si[tid + s];
      }
    }
    __syncthreads();
  }
  int sel = si[0];

  float* prow = atom_probs + (((size_t)b * HEADS_ + head) * L_ + j) * A_;
  for (int a = tid; a < A_; a += 256) prow[a] = (a == sel) ? 1.f : 0.f;
  if (tid == 0) {
    mrow[sel] = 0;
    ind[((size_t)head * L_ + j) * B_ + b] = sel;
  }
}

// cur_in(bf16) = cls + ae_w[selected_atom]
__global__ void curin_kernel(const float* __restrict__ cls,
                             const float* __restrict__ ae_w,
                             const int* __restrict__ ind,
                             unsigned short* __restrict__ curin_bf,
                             int head, int j) {
  int t = blockIdx.x * blockDim.x + threadIdx.x;
  if (t >= B_ * H_) return;
  int b = t / H_, k = t - b * H_;
  int sel = ind[((size_t)head * L_ + j) * B_ + b];
  curin_bf[t] = f2bf(cls[t] + ae_w[(size_t)sel * H_ + k]);
}

// Consequent estimator: pooled is mean of one-hots -> sparse gather of 4 rows.
__global__ void ce_kernel(const int* __restrict__ ind,
                          const float* __restrict__ Wmu, const float* __restrict__ bmu,
                          const float* __restrict__ Wcov, const float* __restrict__ bcov,
                          const float* __restrict__ alpha,
                          float* __restrict__ matcp, int head) {
  int b = blockIdx.x * blockDim.x + threadIdx.x;
  if (b >= B_) return;
  float m[C_];
  #pragma unroll
  for (int c = 0; c < C_; ++c) m[c] = bmu[c];
  float cov = bcov[0];
  for (int j = 0; j < L_; ++j) {
    int s = ind[((size_t)head * L_ + j) * B_ + b];
    #pragma unroll
    for (int c = 0; c < C_; ++c) m[c] += 0.25f * Wmu[(size_t)s * C_ + c];
    cov += 0.25f * Wcov[s];
  }
  float cv = 1.f / (1.f + expf(-cov));
  float n = cv * NDATA_;
  float sf = alpha[0] / n;
  #pragma unroll
  for (int c = 0; c < C_; ++c) {
    float mu = 1.f / (1.f + expf(-m[c]));
    matcp[((size_t)b * HEADS_ + head) * C_ + c] = (mu + sf) / (1.f + 2.f * sf);
  }
}

__global__ void final_kernel(const float* __restrict__ matcp, float* __restrict__ out) {
  int t = blockIdx.x * blockDim.x + threadIdx.x;
  if (t >= B_ * C_) return;
  int b = t / C_, c = t - b * C_;
  float s = 0.f;
  for (int h = 0; h < HEADS_; ++h) s += matcp[((size_t)b * HEADS_ + h) * C_ + c];
  out[t] = logf(s / (float)HEADS_);
}

__global__ void zero_f32_kernel(float* p, int n) {
  int t = blockIdx.x * blockDim.x + threadIdx.x;
  if (t < n) p[t] = 0.f;
}

__global__ void zero_u16_kernel(unsigned short* p, int n) {
  int t = blockIdx.x * blockDim.x + threadIdx.x;
  if (t < n) p[t] = 0;
}

__global__ void copy_i32_kernel(int* __restrict__ dst, const int* __restrict__ src, int n) {
  int t = blockIdx.x * blockDim.x + threadIdx.x;
  if (t < n) dst[t] = src[t];
}

// ---------------------------------------------------------------------------
// Launch: full pipeline on `stream`. Workspace carved from d_ws.
// ---------------------------------------------------------------------------
extern "C" void kernel_launch(void* const* d_in, const int* in_sizes, int n_in,
                              void* d_out, int out_size, void* d_ws, size_t ws_size,
                              hipStream_t stream) {
  (void)in_sizes; (void)n_in; (void)out_size; (void)ws_size;
  const float* x      = (const float*)d_in[0];
  const int*   x_mask = (const int*)d_in[1];
  const float* W1 = (const float*)d_in[2];  const float* b1 = (const float*)d_in[3];
  const float* W2 = (const float*)d_in[4];  const float* b2 = (const float*)d_in[5];
  const float* W3 = (const float*)d_in[6];  const float* b3 = (const float*)d_in[7];
  const float* gWih = (const float*)d_in[8];
  const float* gWhh = (const float*)d_in[9];
  const float* gbih = (const float*)d_in[10];
  const float* gbhh = (const float*)d_in[11];
  const float* headW = (const float*)d_in[12];
  const float* headb = (const float*)d_in[13];
  const float* ae_w  = (const float*)d_in[14];
  const float* ceWmu = (const float*)d_in[15];
  const float* cebmu = (const float*)d_in[16];
  const float* ceWcov = (const float*)d_in[17];
  const float* cebcov = (const float*)d_in[18];
  const float* alpha = (const float*)d_in[19];

  // d_out = out [B,C] ++ atom_probs [B,HEADS,L,A] ++ mat_cp [B,HEADS,C]
  float* out = (float*)d_out;
  float* atom_probs = out + (size_t)B_ * C_;
  float* matcp = atom_probs + (size_t)B_ * HEADS_ * L_ * A_;

  // ---- workspace carving (256B aligned chunks) ----
  char* wsb = (char*)d_ws;
  size_t off = 0;
  auto alloc = [&](size_t bytes) -> void* {
    off = (off + 255) & ~(size_t)255;
    void* p = wsb + off;
    off += bytes;
    return p;
  };
  // fp32 buffers
  float* cls    = (float*)alloc((size_t)B_ * H_ * 4);
  float* hstate = (float*)alloc((size_t)B_ * H_ * 4);
  float* gi     = (float*)alloc((size_t)B_ * 3 * H_ * 4);
  float* gh     = (float*)alloc((size_t)B_ * 3 * H_ * 4);
  float* logits = (float*)alloc((size_t)B_ * A_ * 4);
  int*   mask   = (int*)alloc((size_t)B_ * A_ * 4);
  int*   ind    = (int*)alloc((size_t)HEADS_ * L_ * B_ * 4);
  // bf16 activation buffers
  unsigned short* x_bf     = (unsigned short*)alloc((size_t)B_ * DIN_ * 2);
  unsigned short* h1_bf    = (unsigned short*)alloc((size_t)B_ * H_ * 2);
  unsigned short* h2_bf    = (unsigned short*)alloc((size_t)B_ * H_ * 2);
  unsigned short* cls_bf   = (unsigned short*)alloc((size_t)B_ * H_ * 2);
  unsigned short* h_bf     = (unsigned short*)alloc((size_t)B_ * H_ * 2);
  unsigned short* curin_bf = (unsigned short*)alloc((size_t)B_ * H_ * 2);
  // bf16 TRANSPOSED weight buffers Wt[NP][K]
  unsigned short* W1t   = (unsigned short*)alloc((size_t)H_ * DIN_ * 2);
  unsigned short* W2t   = (unsigned short*)alloc((size_t)H_ * H_ * 2);
  unsigned short* W3t   = (unsigned short*)alloc((size_t)H_ * H_ * 2);
  unsigned short* Wiht  = (unsigned short*)alloc((size_t)HEADS_ * 3 * H_ * H_ * 2);
  unsigned short* Whht  = (unsigned short*)alloc((size_t)HEADS_ * 3 * H_ * H_ * 2);
  unsigned short* headWt = (unsigned short*)alloc((size_t)HEADS_ * AP_ * H_ * 2);

  dim3 blk(256);
  auto cvt_t = [&](const float* s, unsigned short* d, int K, int N, int NP) {
    dim3 grid((K + 31) / 32, (NP + 31) / 32);
    cvt_transpose_kernel<<<grid, blk, 0, stream>>>(s, d, K, N, NP);
  };
  auto gemm = [&](int M, int N, int K, const unsigned short* Ap,
                  const unsigned short* Bp, int NP, const float* biasp,
                  float* Cf, unsigned short* Cb, int act) {
    dim3 grid((N + 63) / 64, (M + 255) / 256);
    gemm_wmma_kernel<<<grid, blk, 0, stream>>>(M, N, K, Ap, Bp, NP, biasp, Cf, Cb, act);
  };

  // ---- one-time conversions: x -> bf16; weights -> transposed bf16 ----
  cvt_bf16_kernel<<<(B_ * DIN_ + 255) / 256, blk, 0, stream>>>(x, x_bf, B_ * DIN_);
  cvt_t(W1, W1t, DIN_, H_, H_);
  cvt_t(W2, W2t, H_, H_, H_);
  cvt_t(W3, W3t, H_, H_, H_);
  for (int h = 0; h < HEADS_; ++h) {
    cvt_t(gWih + (size_t)h * H_ * 3 * H_, Wiht + (size_t)h * 3 * H_ * H_, H_, 3 * H_, 3 * H_);
    cvt_t(gWhh + (size_t)h * H_ * 3 * H_, Whht + (size_t)h * 3 * H_ * H_, H_, 3 * H_, 3 * H_);
    cvt_t(headW + (size_t)h * H_ * A_, headWt + (size_t)h * AP_ * H_, H_, A_, AP_);
  }

  // ---- MLP: cls embedding ----
  gemm(B_, H_, DIN_, x_bf, W1t, H_, b1, nullptr, h1_bf, 1);
  gemm(B_, H_, H_,   h1_bf, W2t, H_, b2, nullptr, h2_bf, 1);
  gemm(B_, H_, H_,   h2_bf, W3t, H_, b3, cls, cls_bf, 0);

  const int nBH = B_ * H_;
  const int nBA = B_ * A_;
  for (int head = 0; head < HEADS_; ++head) {
    copy_i32_kernel<<<(nBA + 255) / 256, blk, 0, stream>>>(mask, x_mask, nBA);
    zero_f32_kernel<<<(nBH + 255) / 256, blk, 0, stream>>>(hstate, nBH);
    zero_u16_kernel<<<(nBH + 255) / 256, blk, 0, stream>>>(h_bf, nBH);
    const unsigned short* Wih = Wiht + (size_t)head * 3 * H_ * H_;
    const unsigned short* Whh = Whht + (size_t)head * 3 * H_ * H_;
    const float* bih = gbih + (size_t)head * 3 * H_;
    const float* bhh = gbhh + (size_t)head * 3 * H_;
    const unsigned short* hW = headWt + (size_t)head * AP_ * H_;
    const float* hb = headb + (size_t)head * A_;
    for (int j = 0; j < L_; ++j) {
      const unsigned short* xin = (j == 0) ? cls_bf : curin_bf;
      gemm(B_, 3 * H_, H_, xin,  Wih, 3 * H_, bih, gi, nullptr, 0);
      gemm(B_, 3 * H_, H_, h_bf, Whh, 3 * H_, bhh, gh, nullptr, 0);
      gru_eltwise_kernel<<<(nBH + 255) / 256, blk, 0, stream>>>(gi, gh, hstate, h_bf, nBH);
      gemm(B_, A_, H_, h_bf, hW, AP_, hb, logits, nullptr, 0);
      select_kernel<<<B_, blk, 0, stream>>>(logits, mask, ind, atom_probs, head, j);
      if (j < L_ - 1)
        curin_kernel<<<(nBH + 255) / 256, blk, 0, stream>>>(cls, ae_w, ind, curin_bf, head, j);
    }
    ce_kernel<<<(B_ + 255) / 256, blk, 0, stream>>>(ind, ceWmu, cebmu, ceWcov,
                                                    cebcov, alpha, matcp, head);
  }
  final_kernel<<<(B_ * C_ + 255) / 256, blk, 0, stream>>>(matcp, out);
}
